// DRL4EVRP_37649683316761
// MI455X (gfx1250) — matrix operand order repaired and checked by reading, not compiled, not run on hardware
//
#include <hip/hip_runtime.h>

// MI455X (gfx1250, wave32) pointer-network decoder.
// B=512, S=100, H=128. distances/slope inputs are unused by the reference.
//
// A_fix/P_fix/static_h stored (b,s,h) (~79 MB total) so they stay L2-resident.
// Per-step GEMMs use v_wmma_f32_16x16x32_bf16; the hot tanh score passes use
// the CDNA5 hardware transcendental v_tanh_f32.

constexpr int B_ = 512;
constexpr int S_ = 100;
constexpr int H_ = 128;

typedef __attribute__((ext_vector_type(16))) __bf16 v16bf;
typedef __attribute__((ext_vector_type(8)))  __bf16 v8bf;
typedef __attribute__((ext_vector_type(8)))  float  v8f;

__device__ __forceinline__ unsigned short f2bfu(float f) {
  union { float f; unsigned u; } v; v.f = f;
  return (unsigned short)((v.u + 0x7FFFu + ((v.u >> 16) & 1u)) >> 16); // RNE
}

// Hardware tanh (TRANS32 op on CDNA5).
__device__ __forceinline__ float fast_tanh(float x) {
#if defined(__HIP_DEVICE_COMPILE__)
# if __has_builtin(__builtin_amdgcn_tanhf)
  return __builtin_amdgcn_tanhf(x);
# elif __has_builtin(__builtin_amdgcn_tanh_f32)
  return __builtin_amdgcn_tanh_f32(x);
# else
  float r;
  asm volatile("v_tanh_f32 %0, %1\n\t"
               "s_delay_alu instid0(TRANS32_DEP_1)"
               : "=v"(r) : "v"(x));
  return r;
# endif
#else
  return tanhf(x);
#endif
}

// A-tile (16x32 bf16): lane L -> row m0+(L&15); lanes>=16 hold K=16..31.
__device__ __forceinline__ v16bf load_a_tile(const unsigned short* src, int ld,
                                             int m0, int k0, int lane) {
  int m   = m0 + (lane & 15);
  int off = (lane >> 4) << 4;                       // 0 or 16
  const __bf16* p = (const __bf16*)src + (size_t)m * ld + k0 + off;
  return *(const v16bf*)p;                          // K = k0+off .. +15
}

// B-tile (32x16 bf16) from weight stored (N x K) row-major:
// lane L -> col n0+(L&15); lanes>=16 hold K=16..31 (contiguous 32B per lane).
__device__ __forceinline__ v16bf load_b_tile(const unsigned short* W, int ldk,
                                             int n0, int k0, int lane) {
  int n    = n0 + (lane & 15);
  int koff = (lane >> 4) << 4;                      // 0 or 16
  return *(const v16bf*)((const __bf16*)W + (size_t)n * ldk + k0 + koff);
}

#define WMMA_BF16(a, b, c) \
  __builtin_amdgcn_wmma_f32_16x16x32_bf16(false, (a), false, (b), (short)0, (c), false, false)

__device__ __forceinline__ float wave_sum32(float p) {
#pragma unroll
  for (int o = 16; o > 0; o >>= 1) p += __shfl_xor(p, o, 32);
  return p;
}

__device__ __forceinline__ float sigmoidf_(float x) {
  return 1.0f / (1.0f + __expf(-x));
}

// ---------------------------------------------------------------- precompute

// bf16 copies of GRU / attention / pointer weight matrices (stored N x K).
__global__ void k_prep_weights(const float* __restrict__ W_ih,
                               const float* __restrict__ W_hh,
                               const float* __restrict__ W_a,
                               const float* __restrict__ W_p,
                               unsigned short* __restrict__ Wih_bf,
                               unsigned short* __restrict__ Whh_bf,
                               unsigned short* __restrict__ Wah_bf,
                               unsigned short* __restrict__ Wpc_bf) {
  int i = blockIdx.x * blockDim.x + threadIdx.x;
  if (i < 3 * H_ * H_) {
    Wih_bf[i] = f2bfu(W_ih[i]);
    Whh_bf[i] = f2bfu(W_hh[i]);
  }
  if (i < H_ * H_) {
    int n = i / H_, k = i % H_;
    Wah_bf[i] = f2bfu(W_a[n * 3 * H_ + 2 * H_ + k]);  // Wa_h = W_a[:, 2H:3H]
    Wpc_bf[i] = f2bfu(W_p[n * 2 * H_ + H_ + k]);      // Wp_c = W_p[:, H:2H]
  }
}

// Collapse rank-2 encoder chains: A_fix/P_fix become K=2 affine maps of the
// raw (static, dynamic) channels. small[]: MAs(256) MAd(256) MPs(256) cA(128) cP(128)
__global__ void k_collapse(const float* __restrict__ W_a,
                           const float* __restrict__ W_p,
                           const float* __restrict__ W_s,
                           const float* __restrict__ W_d,
                           const float* __restrict__ b_s,
                           const float* __restrict__ b_d,
                           float* __restrict__ small) {
  int h = threadIdx.x;
  float mas0 = 0, mas1 = 0, mad0 = 0, mad1 = 0, mps0 = 0, mps1 = 0, ca = 0, cp = 0;
  for (int k = 0; k < H_; ++k) {
    float was = W_a[h * 3 * H_ + k];
    float wad = W_a[h * 3 * H_ + H_ + k];
    float wps = W_p[h * 2 * H_ + k];
    float ws0 = W_s[k * 2 + 0], ws1 = W_s[k * 2 + 1];
    float wd0 = W_d[k * 2 + 0], wd1 = W_d[k * 2 + 1];
    mas0 += was * ws0; mas1 += was * ws1;
    mad0 += wad * wd0; mad1 += wad * wd1;
    mps0 += wps * ws0; mps1 += wps * ws1;
    ca   += was * b_s[k] + wad * b_d[k];
    cp   += wps * b_s[k];
  }
  small[h * 2 + 0] = mas0;        small[h * 2 + 1] = mas1;
  small[256 + h * 2 + 0] = mad0;  small[256 + h * 2 + 1] = mad1;
  small[512 + h * 2 + 0] = mps0;  small[512 + h * 2 + 1] = mps1;
  small[768 + h] = ca;
  small[896 + h] = cp;
}

// static_h / A_fix / P_fix stored (b, s, h), h contiguous.
__global__ void __launch_bounds__(128) k_fix(const float* __restrict__ st,
                                             const float* __restrict__ dy,
                                             const float* __restrict__ W_s,
                                             const float* __restrict__ b_s,
                                             const float* __restrict__ small,
                                             float* __restrict__ A_fix,
                                             float* __restrict__ P_fix,
                                             float* __restrict__ s_h) {
  int s = blockIdx.x, b = blockIdx.y, h = threadIdx.x;
  float st0 = st[b * 2 * S_ + s],       st1 = st[b * 2 * S_ + S_ + s];
  float dy0 = dy[b * 2 * S_ + s],       dy1 = dy[b * 2 * S_ + S_ + s];
  size_t o = ((size_t)b * S_ + s) * H_ + h;
  s_h[o]   = W_s[h * 2] * st0 + W_s[h * 2 + 1] * st1 + b_s[h];
  A_fix[o] = small[h * 2] * st0 + small[h * 2 + 1] * st1 +
             small[256 + h * 2] * dy0 + small[256 + h * 2 + 1] * dy1 + small[768 + h];
  P_fix[o] = small[512 + h * 2] * st0 + small[512 + h * 2 + 1] * st1 + small[896 + h];
}

// h0 = 0; x0 = W_dec @ static[:, :, 0] + b_dec (bf16)
__global__ void __launch_bounds__(128) k_init_state(const float* __restrict__ st,
                                                    const float* __restrict__ W_dec,
                                                    const float* __restrict__ b_dec,
                                                    float* __restrict__ h_f0,
                                                    unsigned short* __restrict__ h_bf0,
                                                    unsigned short* __restrict__ x_bf) {
  int b = blockIdx.x, h = threadIdx.x;
  h_f0[b * H_ + h]  = 0.0f;
  h_bf0[b * H_ + h] = f2bfu(0.0f);
  float s0 = st[b * 2 * S_ + 0], s1 = st[b * 2 * S_ + S_ + 0];
  x_bf[b * H_ + h] = f2bfu(W_dec[h * 2] * s0 + W_dec[h * 2 + 1] * s1 + b_dec[h]);
}

// ---------------------------------------------------------------- per step

// Fused GRU cell + q_a = h_new @ Wa_h^T.
// grid = B/16 blocks x 256 threads (8 waves). Wave w owns output cols 16w..16w+15.
__global__ void __launch_bounds__(256) k_gru(const unsigned short* __restrict__ x_bf,
                                             const unsigned short* __restrict__ h_bf_in,
                                             const float* __restrict__ h_f_in,
                                             unsigned short* __restrict__ h_bf_out,
                                             float* __restrict__ h_f_out,
                                             float* __restrict__ q_a,
                                             const unsigned short* __restrict__ Wih,
                                             const unsigned short* __restrict__ Whh,
                                             const unsigned short* __restrict__ Wah,
                                             const float* __restrict__ b_ih,
                                             const float* __restrict__ b_hh) {
  __shared__ unsigned short lds_h[16 * H_];        // h_new tile (16 batches x H) bf16
  int lane = threadIdx.x & 31;
  int wave = threadIdx.x >> 5;
  int m0 = blockIdx.x * 16;
  int j0 = wave * 16;

  v8f gir = {0,0,0,0,0,0,0,0}, giz = gir, gin = gir;
  v8f ghr = gir, ghz = gir, ghn = gir;
#pragma unroll
  for (int kk = 0; kk < 4; ++kk) {
    int k0 = kk * 32;
    v16bf ax = load_a_tile(x_bf,    H_, m0, k0, lane);
    v16bf ah = load_a_tile(h_bf_in, H_, m0, k0, lane);
    v16bf br = load_b_tile(Wih, H_, j0,          k0, lane);
    v16bf bz = load_b_tile(Wih, H_, H_ + j0,     k0, lane);
    v16bf bn = load_b_tile(Wih, H_, 2 * H_ + j0, k0, lane);
    gir = WMMA_BF16(ax, br, gir);
    giz = WMMA_BF16(ax, bz, giz);
    gin = WMMA_BF16(ax, bn, gin);
    br = load_b_tile(Whh, H_, j0,          k0, lane);
    bz = load_b_tile(Whh, H_, H_ + j0,     k0, lane);
    bn = load_b_tile(Whh, H_, 2 * H_ + j0, k0, lane);
    ghr = WMMA_BF16(ah, br, ghr);
    ghz = WMMA_BF16(ah, bz, ghz);
    ghn = WMMA_BF16(ah, bn, ghn);
  }
  int ncol = j0 + (lane & 15);
  float bir = b_ih[ncol], biz = b_ih[H_ + ncol], bin = b_ih[2 * H_ + ncol];
  float bhr = b_hh[ncol], bhz = b_hh[H_ + ncol], bhn = b_hh[2 * H_ + ncol];
  int mbase = m0 + ((lane >> 4) << 3);             // C/D rows: M = r + 8*(lane>=16)
#pragma unroll
  for (int r = 0; r < 8; ++r) {
    int m = mbase + r;
    float rg = sigmoidf_(gir[r] + bir + ghr[r] + bhr);
    float zg = sigmoidf_(giz[r] + biz + ghz[r] + bhz);
    float ng = fast_tanh(gin[r] + bin + rg * (ghn[r] + bhn));
    float hold = h_f_in[m * H_ + ncol];
    float hn = (1.0f - zg) * ng + zg * hold;
    h_f_out[m * H_ + ncol] = hn;
    unsigned short hb = f2bfu(hn);
    h_bf_out[m * H_ + ncol] = hb;
    lds_h[(m - m0) * H_ + ncol] = hb;
  }
  __syncthreads();
  v8f acc = {0,0,0,0,0,0,0,0};
#pragma unroll
  for (int kk = 0; kk < 4; ++kk) {
    int k0 = kk * 32;
    v16bf ah = load_a_tile(lds_h, H_, 0, k0, lane);
    v16bf bw = load_b_tile(Wah, H_, j0, k0, lane);
    acc = WMMA_BF16(ah, bw, acc);
  }
#pragma unroll
  for (int r = 0; r < 8; ++r)
    q_a[(mbase + r) * H_ + ncol] = acc[r];
}

// Attention: score_s = v_a . tanh(A_fix[b,s,:] + q_a[b,:]); softmax over s;
// context = sum_s attn_s * static_h[b,s,:]. One block (128 thr) per batch.
// Score loop: one b128 load / lane / s; v_a, q_a preloaded in registers.
__global__ void __launch_bounds__(128) k_attn(const float* __restrict__ A_fix,
                                              const float* __restrict__ s_h,
                                              const float* __restrict__ q_a,
                                              const float* __restrict__ v_a,
                                              float* __restrict__ ctx_f,
                                              unsigned short* __restrict__ ctx_bf) {
  int b = blockIdx.x;
  int tid = threadIdx.x, lane = tid & 31, wave = tid >> 5;
  __shared__ float sc[S_];
  __shared__ float red[128];
  __shared__ float4 part[4][32];
  const float4* Ab4 = (const float4*)(A_fix + (size_t)b * S_ * H_);
  float4 va = ((const float4*)v_a)[lane];
  float4 qa = ((const float4*)(q_a + (size_t)b * H_))[lane];
  for (int s = wave; s < S_; s += 4) {
    float4 a = Ab4[s * (H_ / 4) + lane];
    float p = va.x * fast_tanh(a.x + qa.x)
            + va.y * fast_tanh(a.y + qa.y)
            + va.z * fast_tanh(a.z + qa.z)
            + va.w * fast_tanh(a.w + qa.w);
    p = wave_sum32(p);
    if (lane == 0) sc[s] = p;
  }
  __syncthreads();
  float v = (tid < S_) ? sc[tid] : -3.402823466e38f;
  red[tid] = v;
  __syncthreads();
  for (int o = 64; o > 0; o >>= 1) { if (tid < o) red[tid] = fmaxf(red[tid], red[tid + o]); __syncthreads(); }
  float mx = red[0];
  __syncthreads();
  float e = (tid < S_) ? __expf(v - mx) : 0.0f;
  red[tid] = e;
  __syncthreads();
  for (int o = 64; o > 0; o >>= 1) { if (tid < o) red[tid] += red[tid + o]; __syncthreads(); }
  float inv = 1.0f / red[0];
  __syncthreads();
  if (tid < S_) sc[tid] = e * inv;
  __syncthreads();
  // context: each wave accumulates a float4 per lane over its s-range.
  const float4* Sb4 = (const float4*)(s_h + (size_t)b * S_ * H_);
  float4 acc = {0.0f, 0.0f, 0.0f, 0.0f};
  for (int s = wave; s < S_; s += 4) {
    float w = sc[s];
    float4 x = Sb4[s * (H_ / 4) + lane];
    acc.x += w * x.x; acc.y += w * x.y; acc.z += w * x.z; acc.w += w * x.w;
  }
  part[wave][lane] = acc;
  __syncthreads();
  if (wave == 0) {
    float4 a0 = part[0][lane], a1 = part[1][lane], a2 = part[2][lane], a3 = part[3][lane];
    float4 r;
    r.x = a0.x + a1.x + a2.x + a3.x;
    r.y = a0.y + a1.y + a2.y + a3.y;
    r.z = a0.z + a1.z + a2.z + a3.z;
    r.w = a0.w + a1.w + a2.w + a3.w;
    ((float4*)(ctx_f + (size_t)b * H_))[lane] = r;
    unsigned short* cb = ctx_bf + (size_t)b * H_ + lane * 4;
    cb[0] = f2bfu(r.x); cb[1] = f2bfu(r.y); cb[2] = f2bfu(r.z); cb[3] = f2bfu(r.w);
  }
}

// q_p = context @ Wp_c^T (WMMA). grid = B/16 x 256.
__global__ void __launch_bounds__(256) k_qp(const unsigned short* __restrict__ ctx_bf,
                                            const unsigned short* __restrict__ Wpc,
                                            float* __restrict__ q_p) {
  int lane = threadIdx.x & 31;
  int wave = threadIdx.x >> 5;
  int m0 = blockIdx.x * 16, j0 = wave * 16;
  v8f acc = {0,0,0,0,0,0,0,0};
#pragma unroll
  for (int kk = 0; kk < 4; ++kk) {
    int k0 = kk * 32;
    v16bf a = load_a_tile(ctx_bf, H_, m0, k0, lane);
    v16bf w = load_b_tile(Wpc, H_, j0, k0, lane);
    acc = WMMA_BF16(a, w, acc);
  }
  int ncol = j0 + (lane & 15);
  int mbase = m0 + ((lane >> 4) << 3);
#pragma unroll
  for (int r = 0; r < 8; ++r)
    q_p[(mbase + r) * H_ + ncol] = acc[r];
}

// Pointer head: logits_s = v_p . tanh(P_fix[b,s,:] + q_p[b,:]); argmax +
// log-softmax; write outputs; prepare next x (bf16). One block per batch.
__global__ void __launch_bounds__(128) k_ptr(const float* __restrict__ P_fix,
                                             const float* __restrict__ q_p,
                                             const float* __restrict__ v_p,
                                             const float* __restrict__ st,
                                             const float* __restrict__ W_dec,
                                             const float* __restrict__ b_dec,
                                             float* __restrict__ out,
                                             unsigned short* __restrict__ x_bf,
                                             int step) {
  int b = blockIdx.x;
  int tid = threadIdx.x, lane = tid & 31, wave = tid >> 5;
  __shared__ float sc[S_];
  __shared__ float red[128];
  __shared__ int redi[128];
  const float4* Pb4 = (const float4*)(P_fix + (size_t)b * S_ * H_);
  float4 vp = ((const float4*)v_p)[lane];
  float4 qp = ((const float4*)(q_p + (size_t)b * H_))[lane];
  for (int s = wave; s < S_; s += 4) {
    float4 a = Pb4[s * (H_ / 4) + lane];
    float p = vp.x * fast_tanh(a.x + qp.x)
            + vp.y * fast_tanh(a.y + qp.y)
            + vp.z * fast_tanh(a.z + qp.z)
            + vp.w * fast_tanh(a.w + qp.w);
    p = wave_sum32(p);
    if (lane == 0) sc[s] = p;
  }
  __syncthreads();
  float v = (tid < S_) ? sc[tid] : -3.402823466e38f;
  red[tid] = v; redi[tid] = tid;
  __syncthreads();
  for (int o = 64; o > 0; o >>= 1) {
    if (tid < o) {
      float a = red[tid], c = red[tid + o];
      int ia = redi[tid], ic = redi[tid + o];
      if (c > a || (c == a && ic < ia)) { red[tid] = c; redi[tid] = ic; }  // first-max tiebreak
    }
    __syncthreads();
  }
  float mx = red[0];
  int ptr = redi[0];
  __syncthreads();
  float e = (tid < S_) ? __expf(v - mx) : 0.0f;
  red[tid] = e;
  __syncthreads();
  for (int o = 64; o > 0; o >>= 1) { if (tid < o) red[tid] += red[tid + o]; __syncthreads(); }
  float lse = mx + __logf(red[0]);
  if (tid == 0) {
    out[b * S_ + step] = (float)ptr;                 // tour_idx.T
    out[B_ * S_ + b * S_ + step] = mx - lse;         // tour_logp.T (logits[ptr]==mx)
  }
  float s0 = st[b * 2 * S_ + ptr];
  float s1 = st[b * 2 * S_ + S_ + ptr];
  float xv = W_dec[tid * 2 + 0] * s0 + W_dec[tid * 2 + 1] * s1 + b_dec[tid];
  x_bf[b * H_ + tid] = f2bfu(xv);
}

// ---------------------------------------------------------------- launch

extern "C" void kernel_launch(void* const* d_in, const int* in_sizes, int n_in,
                              void* d_out, int out_size, void* d_ws, size_t ws_size,
                              hipStream_t stream) {
  (void)in_sizes; (void)n_in; (void)out_size; (void)ws_size;
  const float* st   = (const float*)d_in[0];   // static  (B,2,S)
  const float* dy   = (const float*)d_in[1];   // dynamic (B,2,S)
  // d_in[2] distances, d_in[3] slope: unused by reference
  const float* W_s   = (const float*)d_in[4];
  const float* b_s   = (const float*)d_in[5];
  const float* W_d   = (const float*)d_in[6];
  const float* b_d   = (const float*)d_in[7];
  const float* W_dec = (const float*)d_in[8];
  const float* b_dec = (const float*)d_in[9];
  const float* W_ih  = (const float*)d_in[10];
  const float* W_hh  = (const float*)d_in[11];
  const float* b_ih  = (const float*)d_in[12];
  const float* b_hh  = (const float*)d_in[13];
  const float* v_a   = (const float*)d_in[14];
  const float* W_a   = (const float*)d_in[15];
  const float* v_p   = (const float*)d_in[16];
  const float* W_p   = (const float*)d_in[17];
  float* out = (float*)d_out;

  const size_t BSH = (size_t)B_ * S_ * H_;     // 6,553,600
  const size_t BH  = (size_t)B_ * H_;          // 65,536
  float* f = (float*)d_ws;
  float* A_fix = f;
  float* P_fix = f + BSH;
  float* s_h   = f + 2 * BSH;
  float* h_f[2]  = { f + 3 * BSH, f + 3 * BSH + BH };
  float* q_a   = f + 3 * BSH + 2 * BH;
  float* q_p   = q_a + BH;
  float* ctx   = q_p + BH;
  float* small = ctx + BH;                     // 1024 floats
  unsigned short* g = (unsigned short*)(small + 1024);   // bf16 region (32B aligned)
  unsigned short* x_bf    = g;
  unsigned short* h_bf[2] = { g + BH, g + 2 * BH };
  unsigned short* ctx_bf  = g + 3 * BH;
  unsigned short* Wih_bf  = g + 4 * BH;
  unsigned short* Whh_bf  = Wih_bf + 3 * H_ * H_;
  unsigned short* Wah_bf  = Whh_bf + 3 * H_ * H_;
  unsigned short* Wpc_bf  = Wah_bf + H_ * H_;
  // total ws: ~80.8 MB

  k_prep_weights<<<(3 * H_ * H_ + 255) / 256, 256, 0, stream>>>(
      W_ih, W_hh, W_a, W_p, Wih_bf, Whh_bf, Wah_bf, Wpc_bf);
  k_collapse<<<1, 128, 0, stream>>>(W_a, W_p, W_s, W_d, b_s, b_d, small);
  k_fix<<<dim3(S_, B_), 128, 0, stream>>>(st, dy, W_s, b_s, small, A_fix, P_fix, s_h);
  k_init_state<<<B_, 128, 0, stream>>>(st, W_dec, b_dec, h_f[0], h_bf[0], x_bf);

  int cur = 0;
  for (int step = 0; step < S_; ++step) {
    int nxt = cur ^ 1;
    k_gru<<<B_ / 16, 256, 0, stream>>>(x_bf, h_bf[cur], h_f[cur],
                                       h_bf[nxt], h_f[nxt], q_a,
                                       Wih_bf, Whh_bf, Wah_bf, b_ih, b_hh);
    k_attn<<<B_, 128, 0, stream>>>(A_fix, s_h, q_a, v_a, ctx, ctx_bf);
    k_qp<<<B_ / 16, 256, 0, stream>>>(ctx_bf, Wpc_bf, q_p);
    k_ptr<<<B_, 128, 0, stream>>>(P_fix, q_p, v_p, st, W_dec, b_dec, out, x_bf, step);
    cur = nxt;
  }
}
